// AttentionLayer_61984968016348
// MI455X (gfx1250) — matrix operand order repaired and checked by reading, not compiled, not run on hardware
//
#include <hip/hip_runtime.h>
#include <hip/hip_bf16.h>
#include <math.h>

// ---------------------------------------------------------------------------
// Problem dims (compile-time)
// ---------------------------------------------------------------------------
#define BH 512
#define W 256
#define C 128
#define C2 256           // 2*C
#define NHEAD 4
#define HD 32            // HEAD_DIM
#define ROWS (BH * W)    // 131072
#define EPS 1e-5f
#define SCALE 0.1f

typedef _Float16 v16h __attribute__((ext_vector_type(16)));
typedef _Float16 v8h  __attribute__((ext_vector_type(8)));
typedef _Float16 v4h  __attribute__((ext_vector_type(4)));
typedef float    v8f  __attribute__((ext_vector_type(8)));
typedef float    v4f  __attribute__((ext_vector_type(4)));

// LDS layout for attention kernel (bytes)
#define SSTR 260                        // padded f32 row stride of score tile
#define SMEM_S_BYTES   (256 * SSTR * 4) // 266240
#define SMEM_KV_OFF    SMEM_S_BYTES
#define SMEM_STAT_OFF  (SMEM_KV_OFF + 32768)
#define SMEM_TOTAL     (SMEM_STAT_OFF + 4 * 256 * 4)   // 303104 < 320KB WGP LDS

static __device__ __forceinline__ float wave_max(float v) {
#pragma unroll
  for (int m = 16; m > 0; m >>= 1) v = fmaxf(v, __shfl_xor(v, m, 32));
  return v;
}
static __device__ __forceinline__ float wave_sum(float v) {
#pragma unroll
  for (int m = 16; m > 0; m >>= 1) v += __shfl_xor(v, m, 32);
  return v;
}
static __device__ __forceinline__ v16h combine(v8h lo, v8h hi8) {
  v16h a;
#pragma unroll
  for (int i = 0; i < 8; ++i) { a[i] = lo[i]; a[8 + i] = hi8[i]; }
  return a;
}
static __device__ __forceinline__ v8f wmma_f16(v16h a, v16h b, v8f c) {
  return __builtin_amdgcn_wmma_f32_16x16x32_f16(false, a, false, b, (short)0, c,
                                                false, false);
}
// memory -> LDS async copy, 16B per lane (ISA §15.18 op 98, ASYNCcnt-tracked)
static __device__ __forceinline__ void async_ld16(const void* gptr,
                                                  void* lptr) {
  unsigned lds = (unsigned)(size_t)lptr;     // addr[31:0] == LDS byte offset
  unsigned long long ga = (unsigned long long)(size_t)gptr;
  asm volatile("global_load_async_to_lds_b128 %0, %1, off"
               :: "v"(lds), "v"(ga) : "memory");
}
static __device__ __forceinline__ void async_wait0() {
  asm volatile("s_wait_asynccnt 0" ::: "memory");
}

// ---------------------------------------------------------------------------
// K0: f32 -> f16 weight conversion
// ---------------------------------------------------------------------------
__global__ __launch_bounds__(256) void k_cvt_f16(const float* __restrict__ src,
                                                 _Float16* __restrict__ dst,
                                                 int n) {
  int i = blockIdx.x * 256 + threadIdx.x;
  if (i < n) dst[i] = (_Float16)src[i];
}

// ---------------------------------------------------------------------------
// K1: LayerNorm (one wave per 128-wide row), output f16
// ---------------------------------------------------------------------------
__global__ __launch_bounds__(256) void k_layernorm(
    const float* __restrict__ x, const float* __restrict__ w,
    const float* __restrict__ b, _Float16* __restrict__ y) {
  const int wave = threadIdx.x >> 5, lane = threadIdx.x & 31;
  const int row = blockIdx.x * 8 + wave;
  if (row >= ROWS) return;
  const float* xr = x + (size_t)row * C;
  v4f xv = *(const v4f*)(xr + lane * 4);
  float s = xv[0] + xv[1] + xv[2] + xv[3];
  float s2 = xv[0] * xv[0] + xv[1] * xv[1] + xv[2] * xv[2] + xv[3] * xv[3];
  s = wave_sum(s);
  s2 = wave_sum(s2);
  const float mu = s * (1.0f / C);
  const float var = s2 * (1.0f / C) - mu * mu;
  const float rstd = rsqrtf(var + EPS);
  v4f wv = *(const v4f*)(w + lane * 4);
  v4f bv = *(const v4f*)(b + lane * 4);
  v4h out;
#pragma unroll
  for (int i = 0; i < 4; ++i)
    out[i] = (_Float16)((xv[i] - mu) * rstd * wv[i] + bv[i]);
  *(v4h*)(y + (size_t)row * C + lane * 4) = out;
}

// ---------------------------------------------------------------------------
// K2: projection GEMM  out[ROWS, 2C] = ln[ROWS, C] @ Wp^T + bias   (WMMA f16)
// ---------------------------------------------------------------------------
__global__ __launch_bounds__(256) void k_proj_gemm(
    const _Float16* __restrict__ A,   // [ROWS, 128]
    const _Float16* __restrict__ Wt,  // [256, 128]  (= Wp, row-major)
    const float* __restrict__ bias,   // [256]
    _Float16* __restrict__ out) {     // [ROWS, 256]
  const int wave = threadIdx.x >> 5, lane = threadIdx.x & 31;
  const int ln_ = lane & 15, hi = lane >> 4;
  const int nb = blockIdx.x * 16;                    // N tile base
  const int mb = (blockIdx.y * 8 + wave) * 16;       // M tile base
  v8f acc = {};
#pragma unroll
  for (int kb = 0; kb < C; kb += 32) {
    const _Float16* ar = A + (size_t)(mb + ln_) * C + kb;
    v16h af = *(const v16h*)(ar + 16 * hi);
    v16h bf = *(const v16h*)(Wt + (size_t)(nb + ln_) * C + kb + 16 * hi);
    acc = wmma_f16(af, bf, acc);
  }
  const float bs = bias[nb + ln_];
#pragma unroll
  for (int r = 0; r < 8; ++r) {
    int m = mb + r + 8 * hi;
    out[(size_t)m * C2 + nb + ln_] = (_Float16)(acc[r] + bs);
  }
}

// ---------------------------------------------------------------------------
// K3: dual-softmax cross attention, one workgroup per (bh, head).
// Score matrix lives entirely in LDS; K tiles (async-copied memory->LDS) are
// overlaid by transposed V tiles once S is built. Both softmax directions'
// AV products use WMMA; each A-fragment feeds two WMMAs (hd 0-15 / 16-31).
// ---------------------------------------------------------------------------
__global__ __launch_bounds__(256) void k_attention(
    const _Float16* __restrict__ kv_l,  // [ROWS, 256]  (k | v)
    const _Float16* __restrict__ kv_r,  // [ROWS, 256]
    _Float16* __restrict__ att_l,       // [ROWS, 128]
    _Float16* __restrict__ att_r) {     // [ROWS, 128]
  extern __shared__ char smem[];
  float* S = (float*)smem;                              // [256][SSTR]
  _Float16* kvh = (_Float16*)(smem + SMEM_KV_OFF);      // 16384 halves
  float* rowmax = (float*)(smem + SMEM_STAT_OFF);
  float* rowsum = rowmax + 256;
  float* colmax = rowsum + 256;
  float* colsum = colmax + 256;

  const int tid = threadIdx.x;
  const int wave = tid >> 5, lane = tid & 31;
  const int ln_ = lane & 15, hi = lane >> 4;
  const int bh = blockIdx.x >> 2, head = blockIdx.x & 3;
  const size_t rowbase = (size_t)bh * W;
  const int hc = head * HD;

  // ---- Phase A: K_l, K_r -> LDS (row-major [256][32]) via async copy -----
  for (int i = tid; i < 1024; i += 256) {
    int pos = i >> 2, ch = (i & 3) * 8;
    async_ld16(kv_l + (rowbase + pos) * C2 + hc + ch, kvh + pos * HD + ch);
    async_ld16(kv_r + (rowbase + pos) * C2 + hc + ch,
               kvh + 8192 + pos * HD + ch);
  }
  async_wait0();
  __syncthreads();

  // ---- Phase B: S = 0.1 * K_l @ K_r^T  (256 tiles of 16x16, K=HD=32) -----
  for (int t = wave; t < 256; t += 8) {
    const int ti = t >> 4, tj = t & 15;
    const _Float16* ar = kvh + (ti * 16 + ln_) * HD;
    v16h af = *(const v16h*)(ar + 16 * hi);
    v16h bf = *(const v16h*)(kvh + 8192 + (tj * 16 + ln_) * HD + 16 * hi);
    v8f acc = {};
    acc = wmma_f16(af, bf, acc);
#pragma unroll
    for (int r = 0; r < 8; ++r)
      S[(ti * 16 + r + 8 * hi) * SSTR + tj * 16 + ln_] = acc[r] * SCALE;
  }
  __syncthreads();

  // ---- Phase C: overlay V (transposed [32][256]) over K; softmax stats ---
  for (int i = tid; i < 1024; i += 256) {
    int pos = i >> 2, ch = (i & 3) * 8;
    v8h vr = *(const v8h*)(kv_r + (rowbase + pos) * C2 + C + hc + ch);
    v8h vl = *(const v8h*)(kv_l + (rowbase + pos) * C2 + C + hc + ch);
#pragma unroll
    for (int j = 0; j < 8; ++j) {
      kvh[(ch + j) * 256 + pos] = vr[j];           // V_r^T
      kvh[8192 + (ch + j) * 256 + pos] = vl[j];    // V_l^T
    }
  }
  for (int rr = 0; rr < 32; ++rr) {   // row stats (softmax over v)
    const int row = wave * 32 + rr;
    float vals[8], mx = -3.4e38f;
#pragma unroll
    for (int it = 0; it < 8; ++it) {
      vals[it] = S[row * SSTR + lane + 32 * it];
      mx = fmaxf(mx, vals[it]);
    }
    mx = wave_max(mx);
    float sm = 0.f;
#pragma unroll
    for (int it = 0; it < 8; ++it) sm += __expf(vals[it] - mx);
    sm = wave_sum(sm);
    if (lane == 0) { rowmax[row] = mx; rowsum[row] = sm; }
  }
  for (int cc = 0; cc < 32; ++cc) {   // column stats (softmax over w)
    const int col = wave * 32 + cc;
    float vals[8], mx = -3.4e38f;
#pragma unroll
    for (int it = 0; it < 8; ++it) {
      vals[it] = S[(lane + 32 * it) * SSTR + col];
      mx = fmaxf(mx, vals[it]);
    }
    mx = wave_max(mx);
    float sm = 0.f;
#pragma unroll
    for (int it = 0; it < 8; ++it) sm += __expf(vals[it] - mx);
    sm = wave_sum(sm);
    if (lane == 0) { colmax[col] = mx; colsum[col] = sm; }
  }
  __syncthreads();

  // ---- Phase D: out_l = softmax_rows(S) @ V_r  (A shared by both N tiles)
  for (int ti = wave; ti < 16; ti += 8) {
    const int m = ti * 16 + ln_;
    const float mx = rowmax[m];
    v8f acc0 = {}, acc1 = {};
    for (int kb = 0; kb < 256; kb += 32) {
      const float* sp = &S[m * SSTR + kb];
      v4f a0 = *(const v4f*)(sp + 16 * hi);
      v4f a1 = *(const v4f*)(sp + 16 * hi + 4);
      v4f a2 = *(const v4f*)(sp + 16 * hi + 8);
      v4f a3 = *(const v4f*)(sp + 16 * hi + 12);
      v16h af;
#pragma unroll
      for (int j = 0; j < 4; ++j) {
        af[j]      = (_Float16)__expf(a0[j] - mx);
        af[4 + j]  = (_Float16)__expf(a1[j] - mx);
        af[8 + j]  = (_Float16)__expf(a2[j] - mx);
        af[12 + j] = (_Float16)__expf(a3[j] - mx);
      }
      v16h b0 = *(const v16h*)(kvh + ln_ * 256 + kb + 16 * hi);
      v16h b1 = *(const v16h*)(kvh + (16 + ln_) * 256 + kb + 16 * hi);
      acc0 = wmma_f16(af, b0, acc0);
      acc1 = wmma_f16(af, b1, acc1);
    }
#pragma unroll
    for (int r = 0; r < 8; ++r) {
      int mr = ti * 16 + r + 8 * hi;
      float inv = 1.0f / rowsum[mr];
      _Float16* o = att_l + (rowbase + mr) * C + hc;
      o[ln_]      = (_Float16)(acc0[r] * inv);
      o[16 + ln_] = (_Float16)(acc1[r] * inv);
    }
  }
  // ---- Phase E: out_r = softmax_cols(S)^T @ V_l --------------------------
  for (int ti = wave; ti < 16; ti += 8) {
    const int m = ti * 16 + ln_;            // output row = right position v
    const float mx = colmax[m];
    v8f acc0 = {}, acc1 = {};
    for (int kb = 0; kb < 256; kb += 32) {  // k = left position w
      v16h af;
#pragma unroll
      for (int e = 0; e < 16; ++e) {
        int kk = kb + 16 * hi + e;
        af[e] = (_Float16)__expf(S[kk * SSTR + m] - mx);   // transposed read
      }
      v16h b0 = *(const v16h*)(kvh + 8192 + ln_ * 256 + kb + 16 * hi);
      v16h b1 = *(const v16h*)(kvh + 8192 + (16 + ln_) * 256 + kb + 16 * hi);
      acc0 = wmma_f16(af, b0, acc0);
      acc1 = wmma_f16(af, b1, acc1);
    }
#pragma unroll
    for (int r = 0; r < 8; ++r) {
      int mr = ti * 16 + r + 8 * hi;
      float inv = 1.0f / colsum[mr];
      _Float16* o = att_r + (rowbase + mr) * C + hc;
      o[ln_]      = (_Float16)(acc0[r] * inv);
      o[16 + ln_] = (_Float16)(acc1[r] * inv);
    }
  }
}

// ---------------------------------------------------------------------------
// K4: output projection + bias + residual, f32 result
// ---------------------------------------------------------------------------
__global__ __launch_bounds__(256) void k_out_gemm(
    const _Float16* __restrict__ A,   // [ROWS, 128] attention output (f16)
    const _Float16* __restrict__ Wt,  // [128, 128]
    const float* __restrict__ bias,   // [128]
    const float* __restrict__ feat,   // [ROWS, 128] residual
    float* __restrict__ out) {        // [ROWS, 128]
  const int wave = threadIdx.x >> 5, lane = threadIdx.x & 31;
  const int ln_ = lane & 15, hi = lane >> 4;
  const int nb = blockIdx.x * 16;
  const int mb = (blockIdx.y * 8 + wave) * 16;
  v8f acc = {};
#pragma unroll
  for (int kb = 0; kb < C; kb += 32) {
    const _Float16* ar = A + (size_t)(mb + ln_) * C + kb;
    v16h af = *(const v16h*)(ar + 16 * hi);
    v16h bf = *(const v16h*)(Wt + (size_t)(nb + ln_) * C + kb + 16 * hi);
    acc = wmma_f16(af, bf, acc);
  }
  const float bs = bias[nb + ln_];
#pragma unroll
  for (int r = 0; r < 8; ++r) {
    size_t idx = (size_t)(mb + r + 8 * hi) * C + nb + ln_;
    out[idx] = acc[r] + bs + feat[idx];
  }
}

// ---------------------------------------------------------------------------
// Launch
// ---------------------------------------------------------------------------
extern "C" void kernel_launch(void* const* d_in, const int* in_sizes, int n_in,
                              void* d_out, int out_size, void* d_ws,
                              size_t ws_size, hipStream_t stream) {
  (void)in_sizes; (void)n_in; (void)out_size; (void)ws_size;
  const float* feat_l  = (const float*)d_in[0];
  const float* feat_r  = (const float*)d_in[1];
  const float* wlp     = (const float*)d_in[2];   // (256,128)
  const float* wrp     = (const float*)d_in[3];
  const float* blp     = (const float*)d_in[4];   // (256,)
  const float* brp     = (const float*)d_in[5];
  const float* wlo     = (const float*)d_in[6];   // (128,128)
  const float* blo     = (const float*)d_in[7];
  const float* wro     = (const float*)d_in[8];
  const float* bro     = (const float*)d_in[9];
  const float* nw      = (const float*)d_in[10];
  const float* nb_     = (const float*)d_in[11];
  float* outL = (float*)d_out;
  float* outR = outL + (size_t)ROWS * C;

  // workspace carve-up (halves)
  _Float16* p = (_Float16*)d_ws;
  _Float16* wlp16 = p;               p += 2 * C * C;        // 32768
  _Float16* wrp16 = p;               p += 2 * C * C;
  _Float16* wlo16 = p;               p += C * C;            // 16384
  _Float16* wro16 = p;               p += C * C;
  _Float16* lnL   = p;               p += (size_t)ROWS * C; // reused as attL
  _Float16* lnR   = p;               p += (size_t)ROWS * C; // reused as attR
  _Float16* kvL   = p;               p += (size_t)ROWS * C2;
  _Float16* kvR   = p;               p += (size_t)ROWS * C2;
  _Float16* attL = lnL;  // ln buffers dead after projection
  _Float16* attR = lnR;

  // K0: weights -> f16
  k_cvt_f16<<<(2 * C * C + 255) / 256, 256, 0, stream>>>(wlp, wlp16, 2 * C * C);
  k_cvt_f16<<<(2 * C * C + 255) / 256, 256, 0, stream>>>(wrp, wrp16, 2 * C * C);
  k_cvt_f16<<<(C * C + 255) / 256, 256, 0, stream>>>(wlo, wlo16, C * C);
  k_cvt_f16<<<(C * C + 255) / 256, 256, 0, stream>>>(wro, wro16, C * C);

  // K1: layernorm
  k_layernorm<<<ROWS / 8, 256, 0, stream>>>(feat_l, nw, nb_, lnL);
  k_layernorm<<<ROWS / 8, 256, 0, stream>>>(feat_r, nw, nb_, lnR);

  // K2: KV projections (N=256)
  dim3 gp(C2 / 16, ROWS / (16 * 8));
  k_proj_gemm<<<gp, 256, 0, stream>>>(lnL, wlp16, blp, kvL);
  k_proj_gemm<<<gp, 256, 0, stream>>>(lnR, wrp16, brp, kvR);

  // K3: LDS-resident dual-softmax attention, one WG per (bh, head)
  k_attention<<<BH * NHEAD, 256, SMEM_TOTAL, stream>>>(kvL, kvR, attL, attR);

  // K4: output projections + residual (N=128)
  dim3 go(C / 16, ROWS / (16 * 8));
  k_out_gemm<<<go, 256, 0, stream>>>(attL, wlo16, blo, feat_l, outL);
  k_out_gemm<<<go, 256, 0, stream>>>(attR, wro16, bro, feat_r, outR);
}
